// OriginChannel_47708496724511
// MI455X (gfx1250) — matrix operand order repaired and checked by reading, not compiled, not run on hardware
//
#include <hip/hip_runtime.h>

#define NN   20000
#define EE   160000
#define GG   800
#define HH   3
#define HID  128
#define DN   44
#define DE   12
#define BN_EPS 1e-5f
#define KEY_NEG_INF 0x007FFFFFu   // fkey(-inf)

typedef __attribute__((ext_vector_type(16))) __bf16 v16bf;
typedef __attribute__((ext_vector_type(8)))  float  v8f;

// ---------------- device helpers ----------------

__device__ __forceinline__ unsigned short bf16bits(float f) {
  unsigned u = __float_as_uint(f);
  return (unsigned short)((u + 0x7FFFu + ((u >> 16) & 1u)) >> 16);  // RNE
}
__device__ __forceinline__ __bf16 f2bf(float f) {
  return __builtin_bit_cast(__bf16, bf16bits(f));
}
__device__ __forceinline__ unsigned fkey(float f) {      // order-preserving float->uint
  unsigned u = __float_as_uint(f);
  return (u & 0x80000000u) ? ~u : (u | 0x80000000u);
}
__device__ __forceinline__ float unfkey(unsigned k) {
  unsigned u = (k & 0x80000000u) ? (k & 0x7FFFFFFFu) : ~k;
  return __uint_as_float(u);
}

// ACT: 0=none, 1=leaky0.01, 2=leaky0.2, 3=relu
template <int ACT>
__device__ __forceinline__ float actf(float v) {
  if (ACT == 1) return v >= 0.f ? v : 0.01f * v;
  if (ACT == 2) return v >= 0.f ? v : 0.2f  * v;
  if (ACT == 3) return fmaxf(v, 0.f);
  return v;
}

// ---------------- WMMA GEMM: Out[M,Nout] = act(A[M,K] @ W[Nout,K]^T + bias) ----------------
// 32x32 output tile per wave (2 A-frags x 2 B-frags -> 4 WMMAs / K-step), 8 waves/block.
// A is f32 (ABF=false) or bf16 (ABF=true); W is f32, converted on the fly (cache-hot).
// Fragment layouts per ISA 7.12.2 (wave32):
//   A 16x32 bf16 : lane<16 row r, K=kb+{0..7,16..23}; lane>=16 row r, K=kb+{8..15,24..31}
//   B 32x16 bf16 : lane<16 col r, K=kb+{0..7,16..23}; lane>=16 col r, K=kb+{8..15,24..31}
//   C/D 16x16 f32: lane col r, elem j -> row hlf*8+j
template <int ACT, bool ABF>
__global__ void __launch_bounds__(256)
gemm_wmma(const void* __restrict__ Ain, int lda,
          const float* __restrict__ W, int ldw,
          const float* __restrict__ bias,
          float* __restrict__ Out, int ldo,
          int M, int K, int Nout)
{
  const int lane   = threadIdx.x & 31;
  const int wave   = threadIdx.x >> 5;
  const int tilesN = Nout >> 5;
  const int tile   = blockIdx.x * 8 + wave;
  const int tileM  = tile / tilesN;
  if (tileM * 32 >= M) return;
  const int tileN = tile - tileM * tilesN;
  const int m0 = tileM << 5, n0 = tileN << 5;
  const int hlf = lane >> 4, r = lane & 15;

  const float*          aF0 = (const float*)Ain + (size_t)(m0 + r) * lda;
  const float*          aF1 = aF0 + (size_t)16 * lda;
  const unsigned short* aB0 = (const unsigned short*)Ain + (size_t)(m0 + r) * lda;
  const unsigned short* aB1 = aB0 + (size_t)16 * lda;
  const float* w0 = W + (size_t)(n0 + r) * ldw + hlf * 8;
  const float* w1 = w0 + (size_t)16 * ldw;

  v8f acc00 = {}, acc01 = {}, acc10 = {}, acc11 = {};
  for (int kb = 0; kb < K; kb += 32) {
    v16bf a0, a1, b0, b1;
    if (ABF) {
      const unsigned short* p0 = aB0 + kb + hlf * 8;
      const unsigned short* p1 = aB1 + kb + hlf * 8;
      __builtin_prefetch(p0 + 32, 0, 1);     // global_prefetch_b8
#pragma unroll
      for (int i = 0; i < 8; ++i) {
        a0[i]     = __builtin_bit_cast(__bf16, p0[i]);
        a0[8 + i] = __builtin_bit_cast(__bf16, p0[16 + i]);
        a1[i]     = __builtin_bit_cast(__bf16, p1[i]);
        a1[8 + i] = __builtin_bit_cast(__bf16, p1[16 + i]);
      }
    } else {
      const float* p0 = aF0 + kb + hlf * 8;
      const float* p1 = aF1 + kb + hlf * 8;
      __builtin_prefetch(p0 + 32, 0, 1);
#pragma unroll
      for (int i = 0; i < 8; ++i) {
        a0[i]     = f2bf(p0[i]);
        a0[8 + i] = f2bf(p0[16 + i]);
        a1[i]     = f2bf(p1[i]);
        a1[8 + i] = f2bf(p1[16 + i]);
      }
    }
    const float* q0 = w0 + kb;
    const float* q1 = w1 + kb;
#pragma unroll
    for (int i = 0; i < 8; ++i) { b0[i] = f2bf(q0[i]); b0[8 + i] = f2bf(q0[16 + i]); b1[i] = f2bf(q1[i]); b1[8 + i] = f2bf(q1[16 + i]); }

    acc00 = __builtin_amdgcn_wmma_f32_16x16x32_bf16(false, a0, false, b0, (short)0, acc00, false, false);
    acc01 = __builtin_amdgcn_wmma_f32_16x16x32_bf16(false, a0, false, b1, (short)0, acc01, false, false);
    acc10 = __builtin_amdgcn_wmma_f32_16x16x32_bf16(false, a1, false, b0, (short)0, acc10, false, false);
    acc11 = __builtin_amdgcn_wmma_f32_16x16x32_bf16(false, a1, false, b1, (short)0, acc11, false, false);
  }
  const float bv0 = bias ? bias[n0 + r]      : 0.f;
  const float bv1 = bias ? bias[n0 + 16 + r] : 0.f;
  float* o0 = Out + (size_t)(m0 + hlf * 8) * ldo + n0 + r;
  float* o1 = o0 + (size_t)16 * ldo;
#pragma unroll
  for (int j = 0; j < 8; ++j) {
    o0[(size_t)j * ldo]      = actf<ACT>(acc00[j] + bv0);
    o0[(size_t)j * ldo + 16] = actf<ACT>(acc01[j] + bv1);
    o1[(size_t)j * ldo]      = actf<ACT>(acc10[j] + bv0);
    o1[(size_t)j * ldo + 16] = actf<ACT>(acc11[j] + bv1);
  }
}

// ---------------- small-K encoder GEMM (K=44 or 12) ----------------
__global__ void gemm_naive(const float* __restrict__ A, const float* __restrict__ W,
                           const float* __restrict__ bias, float* __restrict__ Out,
                           int M, int K)
{
  int idx = blockIdx.x * blockDim.x + threadIdx.x;
  if (idx >= M * HID) return;
  int row = idx >> 7, col = idx & (HID - 1);
  const float* a = A + (size_t)row * K;
  const float* w = W + (size_t)col * K;
  float s = bias[col];
  for (int k = 0; k < K; ++k) s += a[k] * w[k];
  Out[idx] = s;
}

// ---------------- BatchNorm (batch stats over rows, 128 cols) ----------------
__global__ void bn_clear(float* sums) { sums[threadIdx.x] = 0.f; }   // <<<1,256>>>

__global__ void bn_partial(const float* __restrict__ X, float* __restrict__ sums, int M)
{
  int c = threadIdx.x;                 // 128 threads
  int r0 = blockIdx.x * 64;
  int r1 = r0 + 64; if (r1 > M) r1 = M;
  float s = 0.f, s2 = 0.f;
  for (int rr = r0; rr < r1; ++rr) {
    float v = X[(size_t)rr * HID + c];
    s += v; s2 += v * v;
  }
  atomicAdd(&sums[c], s);
  atomicAdd(&sums[HID + c], s2);
}

// OBF=true writes bf16 (feeds WMMA GEMM A-side directly, halves HBM traffic)
template <int ACT, bool OBF>
__global__ void bn_norm(const float* __restrict__ Xin, void* __restrict__ Xout,
                        const float* __restrict__ sums,
                        const float* __restrict__ gma, const float* __restrict__ beta, int M)
{
  int idx = blockIdx.x * blockDim.x + threadIdx.x;
  if (idx >= M * HID) return;
  int c = idx & (HID - 1);
  float mu  = sums[c] / (float)M;
  float var = sums[HID + c] / (float)M - mu * mu;     // biased var
  float inv = rsqrtf(var + BN_EPS);
  float v = actf<ACT>((Xin[idx] - mu) * inv * gma[c] + beta[c]);
  if (OBF) ((unsigned short*)Xout)[idx] = bf16bits(v);
  else     ((float*)Xout)[idx] = v;
}

// ---------------- per-row dot with a 128-vector (one wave per row) ----------------
__global__ void rowdot(const float* __restrict__ Xr, const float* __restrict__ v,
                       float* __restrict__ out, int M)
{
  int gid = blockIdx.x * blockDim.x + threadIdx.x;
  int row = gid >> 5, lane = threadIdx.x & 31;
  if (row >= M) return;
  const float* xr = Xr + (size_t)row * HID;
  float s = 0.f;
  for (int i = lane; i < HID; i += 32) s += xr[i] * v[i];
  for (int o = 16; o; o >>= 1) s += __shfl_xor(s, o, 32);
  if (lane == 0) out[row] = s;
}

// ---------------- GATEConv edge kernels ----------------
// m[e] = leaky( xg[src[e]] + eg[e], 0.2 )   (in place on eg/m buffer)
__global__ void edge_m_fuse(const float* __restrict__ xg, float* __restrict__ m,
                            const int* __restrict__ src)
{
  int idx = blockIdx.x * blockDim.x + threadIdx.x;
  int e = idx >> 7;
  if (e >= EE) return;
  int c = idx & (HID - 1);
  float v = xg[(size_t)src[e] * HID + c] + m[idx];
  m[idx] = v >= 0.f ? v : 0.2f * v;
}

// alpha[e] = leaky(m[e].gattl + xr[dst[e]], 0.2); atomicMax segment max (wave per edge)
__global__ void gate_alpha_max(const float* __restrict__ m, const float* __restrict__ attl,
                               const float* __restrict__ xr, const int* __restrict__ dst,
                               float* __restrict__ alpha, unsigned* __restrict__ smax)
{
  int gid = blockIdx.x * blockDim.x + threadIdx.x;
  int e = gid >> 5, lane = threadIdx.x & 31;
  if (e >= EE) return;
  const float* mr = m + (size_t)e * HID;
  float s = 0.f;
  for (int i = lane; i < HID; i += 32) s += mr[i] * attl[i];
  for (int o = 16; o; o >>= 1) s += __shfl_xor(s, o, 32);
  if (lane == 0) {
    int d = dst[e];
    float a = s + xr[d];
    a = a >= 0.f ? a : 0.2f * a;
    alpha[e] = a;
    atomicMax(&smax[d], fkey(a));
  }
}

// GATConv: alpha[e] = leaky(sdot[src]+ddot[dst], 0.01)
__global__ void gat_alpha_max(const float* __restrict__ sdot, const float* __restrict__ ddot,
                              const int* __restrict__ src, const int* __restrict__ dst,
                              float* __restrict__ alpha, unsigned* __restrict__ smax)
{
  int e = blockIdx.x * blockDim.x + threadIdx.x;
  if (e >= EE) return;
  float a = sdot[src[e]] + ddot[dst[e]];
  a = a >= 0.f ? a : 0.01f * a;
  alpha[e] = a;
  atomicMax(&smax[dst[e]], fkey(a));
}

// mol readout: alpha[r] = leaky(asrc[r] + gdot[batch[r]], 0.01)
__global__ void mol_alpha_max(const float* __restrict__ asrc, const float* __restrict__ gdot,
                              const int* __restrict__ batch,
                              float* __restrict__ alpha, unsigned* __restrict__ smax)
{
  int rr = blockIdx.x * blockDim.x + threadIdx.x;
  if (rr >= NN) return;
  float a = asrc[rr] + gdot[batch[rr]];
  a = a >= 0.f ? a : 0.01f * a;
  alpha[rr] = a;
  atomicMax(&smax[batch[rr]], fkey(a));
}

// exp(a - segmax), accumulate denominator; alpha overwritten with exp
__global__ void seg_exp_sum(float* __restrict__ alpha, const unsigned* __restrict__ smax,
                            float* __restrict__ den, const int* __restrict__ seg, int n)
{
  int i = blockIdx.x * blockDim.x + threadIdx.x;
  if (i >= n) return;
  int s = seg[i];
  float mv = unfkey(smax[s]);
  if (!(mv > -3.0e38f && mv < 3.0e38f)) mv = 0.f;
  float ex = __expf(alpha[i] - mv);
  alpha[i] = ex;
  atomicAdd(&den[s], ex);
}

// out[seg[i]] += (alpha[i]/(den[seg[i]]+1e-16)) * Rows[gather?gather[i]:i]
__global__ void seg_scatter(const float* __restrict__ Rows, const int* __restrict__ gather,
                            const float* __restrict__ alpha, const float* __restrict__ den,
                            const int* __restrict__ seg, float* __restrict__ out, int n)
{
  int idx = blockIdx.x * blockDim.x + threadIdx.x;
  int i = idx >> 7;
  if (i >= n) return;
  int c = idx & (HID - 1);
  int s = seg[i];
  int rs = gather ? gather[i] : i;
  float w = alpha[i] / (den[s] + 1e-16f);
  atomicAdd(&out[(size_t)s * HID + c], w * Rows[(size_t)rs * HID + c]);
}

// g[batch[r]] += x[r]
__global__ void seg_sum_rows(const float* __restrict__ X, const int* __restrict__ batch,
                             float* __restrict__ g, int M)
{
  int idx = blockIdx.x * blockDim.x + threadIdx.x;
  int rr = idx >> 7;
  if (rr >= M) return;
  int c = idx & (HID - 1);
  atomicAdd(&g[(size_t)batch[rr] * HID + c], X[idx]);
}

// GRU gate fusion: hx = relu((1-z)*n + z*hx), gi/gh are [M,384]
__global__ void gru_fuse(const float* __restrict__ gi, const float* __restrict__ gh,
                         float* __restrict__ hx, int M)
{
  int idx = blockIdx.x * blockDim.x + threadIdx.x;
  int row = idx >> 7;
  if (row >= M) return;
  int c = idx & (HID - 1);
  const float* a = gi + (size_t)row * 384;
  const float* b = gh + (size_t)row * 384;
  float rg = 1.f / (1.f + __expf(-(a[c] + b[c])));
  float zg = 1.f / (1.f + __expf(-(a[HID + c] + b[HID + c])));
  float ng = tanhf(a[2 * HID + c] + rg * b[2 * HID + c]);
  float v = (1.f - zg) * ng + zg * hx[idx];
  hx[idx] = fmaxf(v, 0.f);
}

__global__ void elu_inplace(float* __restrict__ x, int n)
{
  int i = blockIdx.x * blockDim.x + threadIdx.x;
  if (i >= n) return;
  float v = x[i];
  x[i] = v > 0.f ? v : __expf(v) - 1.f;
}
__global__ void relu_inplace(float* __restrict__ x, int n)
{
  int i = blockIdx.x * blockDim.x + threadIdx.x;
  if (i < n) x[i] = fmaxf(x[i], 0.f);
}
__global__ void fill_f32(float* __restrict__ p, float v, int n)
{
  int i = blockIdx.x * blockDim.x + threadIdx.x;
  if (i < n) p[i] = v;
}
__global__ void fill_u32(unsigned* __restrict__ p, unsigned v, int n)
{
  int i = blockIdx.x * blockDim.x + threadIdx.x;
  if (i < n) p[i] = v;
}
__global__ void fill_rowbias(float* __restrict__ out, const float* __restrict__ bias, int rows)
{
  int idx = blockIdx.x * blockDim.x + threadIdx.x;
  if (idx < rows * HID) out[idx] = bias[idx & (HID - 1)];
}

// ---------------- host orchestration ----------------

static inline int cdiv(int a, int b) { return (a + b - 1) / b; }

extern "C" void kernel_launch(void* const* d_in, const int* in_sizes, int n_in,
                              void* d_out, int out_size, void* d_ws, size_t ws_size,
                              hipStream_t stream)
{
  (void)in_sizes; (void)n_in; (void)out_size; (void)ws_size;

  const float* X       = (const float*)d_in[0];
  const float* EA      = (const float*)d_in[1];
  const float* node_w  = (const float*)d_in[2];
  const float* node_b  = (const float*)d_in[3];
  const float* node_bg = (const float*)d_in[4];
  const float* node_bb = (const float*)d_in[5];
  const float* edge_w  = (const float*)d_in[6];
  const float* edge_b  = (const float*)d_in[7];
  const float* edge_bg = (const float*)d_in[8];
  const float* edge_bb = (const float*)d_in[9];
  const float* lin1_w  = (const float*)d_in[10];
  const float* lin1_b  = (const float*)d_in[11];
  const float* gl1_w   = (const float*)d_in[12];   // [H,128,256]
  const float* gl2_w   = (const float*)d_in[13];   // [H,128,128]
  const float* gattL   = (const float*)d_in[14];
  const float* gattR   = (const float*)d_in[15];
  const float* gbias   = (const float*)d_in[16];
  const float* conv_w  = (const float*)d_in[17];   // [H,2,128,128]
  const float* conv_as = (const float*)d_in[18];
  const float* conv_ad = (const float*)d_in[19];
  const float* conv_b  = (const float*)d_in[20];
  const float* gwih    = (const float*)d_in[21];   // [H,3,384,128]
  const float* gwhh    = (const float*)d_in[22];
  const float* gbih    = (const float*)d_in[23];   // [H,3,384]
  const float* gbhh    = (const float*)d_in[24];
  const float* mws     = (const float*)d_in[25];
  const float* mwd     = (const float*)d_in[26];
  const float* mas     = (const float*)d_in[27];
  const float* mad     = (const float*)d_in[28];
  const float* mbias   = (const float*)d_in[29];
  const float* mgwih   = (const float*)d_in[30];   // [H,384,128]
  const float* mgwhh   = (const float*)d_in[31];
  const float* mgbih   = (const float*)d_in[32];
  const float* mgbhh   = (const float*)d_in[33];
  const float* out_w   = (const float*)d_in[34];
  const float* out_b   = (const float*)d_in[35];
  const float* att_w   = (const float*)d_in[36];   // [128,384]
  const float* att_b   = (const float*)d_in[37];
  const float* att_bg  = (const float*)d_in[38];
  const float* att_bb  = (const float*)d_in[39];
  const int*   eidx    = (const int*)d_in[40];
  const int*   batch   = (const int*)d_in[41];
  const int* src = eidx;
  const int* dst = eidx + EE;
  float* OUT = (float*)d_out;

  // workspace carve-up (float units)
  float* F = (float*)d_ws;
  size_t off = 0;
  auto alloc = [&](size_t n) { float* p = F + off; off += n; return p; };
  float* exb  = alloc((size_t)NN * HID);            // node-encoder pre-BN scratch
  float* eeb  = alloc((size_t)EE * HID);            // edge-encoder pre-BN scratch
  unsigned short* exbf = (unsigned short*)alloc((size_t)NN * HID / 2);  // bf16 encoded nodes
  unsigned short* eebf = (unsigned short*)alloc((size_t)EE * HID / 2);  // bf16 encoded edges
  float* xb   = alloc((size_t)NN * HID);   // head node state
  float* xgb  = alloc((size_t)NN * HID);   // x@gl1a / xw / g@mol_wd
  float* mbuf = alloc((size_t)EE * HID);   // eg / m (edge feature msgs)
  float* amb  = alloc((size_t)NN * HID);   // aggregated msgs / conv h / mol h
  float* hbb  = alloc((size_t)NN * HID);   // GATEConv h (elu'd in place)
  float* gib  = alloc((size_t)NN * 384);
  float* ghb  = alloc((size_t)NN * 384);
  float* alp  = alloc((size_t)EE);         // per-edge (or per-node) alpha
  unsigned* smax = (unsigned*)alloc((size_t)NN);
  float* sden = alloc((size_t)NN);
  float* dA   = alloc((size_t)NN);         // xr / sdot / a_src
  float* dB   = alloc((size_t)NN);         // ddot
  float* dG   = alloc((size_t)GG);         // gdot
  float* gbuf = alloc((size_t)GG * HID);
  float* xsb  = alloc((size_t)NN * HID);
  float* catb = alloc((size_t)GG * 384);
  float* ftmp = alloc((size_t)GG * HID);
  float* bns  = alloc(256);

  const dim3 B256(256);
  auto gemmF = [&](const float* A, int lda, const float* W, int ldw, const float* bias,
                   float* O, int ldo, int M, int K, int Nout, int act) {
    int blocks = cdiv((M / 32) * (Nout / 32), 8);
    if (act == 1) gemm_wmma<1, false><<<blocks, B256, 0, stream>>>(A, lda, W, ldw, bias, O, ldo, M, K, Nout);
    else          gemm_wmma<0, false><<<blocks, B256, 0, stream>>>(A, lda, W, ldw, bias, O, ldo, M, K, Nout);
  };
  auto gemmB = [&](const unsigned short* A, int lda, const float* W, int ldw, const float* bias,
                   float* O, int ldo, int M, int K, int Nout, int act) {
    int blocks = cdiv((M / 32) * (Nout / 32), 8);
    if (act == 1) gemm_wmma<1, true><<<blocks, B256, 0, stream>>>(A, lda, W, ldw, bias, O, ldo, M, K, Nout);
    else          gemm_wmma<0, true><<<blocks, B256, 0, stream>>>(A, lda, W, ldw, bias, O, ldo, M, K, Nout);
  };
  // BN with batch stats; encoders emit bf16 + leaky(0.01), final emits f32 + relu
  auto bn = [&](const float* Xin, void* Xout, const float* g, const float* b, int M, bool enc) {
    bn_clear<<<1, 256, 0, stream>>>(bns);
    bn_partial<<<cdiv(M, 64), dim3(128), 0, stream>>>(Xin, bns, M);
    if (enc) bn_norm<1, true ><<<cdiv(M * HID, 256), B256, 0, stream>>>(Xin, Xout, bns, g, b, M);
    else     bn_norm<3, false><<<cdiv(M * HID, 256), B256, 0, stream>>>(Xin, Xout, bns, g, b, M);
  };

  // ---- encoders + BN + leaky(0.01), stored bf16 for WMMA A-side ----
  gemm_naive<<<cdiv(NN * HID, 256), B256, 0, stream>>>(X, node_w, node_b, exb, NN, DN);
  bn(exb, exbf, node_bg, node_bb, NN, true);
  gemm_naive<<<cdiv(EE * HID, 256), B256, 0, stream>>>(EA, edge_w, edge_b, eeb, EE, DE);
  bn(eeb, eebf, edge_bg, edge_bb, EE, true);

  for (int h = 0; h < HH; ++h) {
    // x = leaky(ex @ lin1.T + b, 0.01)
    gemmB(exbf, HID, lin1_w + (size_t)h * HID * HID, HID, lin1_b + h * HID, xb, HID, NN, HID, HID, 1);

    // ---- GATEConv ----
    gemmF(xb, HID, gl1_w + (size_t)h * HID * 2 * HID, 2 * HID, nullptr, xgb, HID, NN, HID, HID, 0);
    gemmB(eebf, HID, gl1_w + (size_t)h * HID * 2 * HID + HID, 2 * HID, nullptr, mbuf, HID, EE, HID, HID, 0);
    edge_m_fuse<<<cdiv(EE * HID, 256), B256, 0, stream>>>(xgb, mbuf, src);
    rowdot<<<cdiv(NN * 32, 256), B256, 0, stream>>>(xb, gattR + h * HID, dA, NN);
    fill_u32<<<cdiv(NN, 256), B256, 0, stream>>>(smax, KEY_NEG_INF, NN);
    gate_alpha_max<<<cdiv(EE * 32, 256), B256, 0, stream>>>(mbuf, gattL + h * HID, dA, dst, alp, smax);
    fill_f32<<<cdiv(NN, 256), B256, 0, stream>>>(sden, 0.f, NN);
    seg_exp_sum<<<cdiv(EE, 256), B256, 0, stream>>>(alp, smax, sden, dst, EE);
    fill_f32<<<cdiv(NN * HID, 256), B256, 0, stream>>>(amb, 0.f, NN * HID);
    seg_scatter<<<cdiv(EE * HID, 256), B256, 0, stream>>>(mbuf, nullptr, alp, sden, dst, amb, EE);
    // h = (sum alpha*m) @ gl2.T + gbias ; x = relu(gru(elu(h), x))
    gemmF(amb, HID, gl2_w + (size_t)h * HID * HID, HID, gbias + h * HID, hbb, HID, NN, HID, HID, 0);
    elu_inplace<<<cdiv(NN * HID, 256), B256, 0, stream>>>(hbb, NN * HID);
    gemmF(hbb, HID, gwih + (size_t)(h * 3) * 384 * HID, HID, gbih + (h * 3) * 384, gib, 384, NN, HID, 384, 0);
    gemmF(xb,  HID, gwhh + (size_t)(h * 3) * 384 * HID, HID, gbhh + (h * 3) * 384, ghb, 384, NN, HID, 384, 0);
    gru_fuse<<<cdiv(NN * HID, 256), B256, 0, stream>>>(gib, ghb, xb, NN);

    // ---- GATConv x2 ----
    for (int l = 0; l < 2; ++l) {
      const size_t cw = (size_t)(h * 2 + l) * HID * HID;
      gemmF(xb, HID, conv_w + cw, HID, nullptr, xgb, HID, NN, HID, HID, 0);
      rowdot<<<cdiv(NN * 32, 256), B256, 0, stream>>>(xgb, conv_as + (h * 2 + l) * HID, dA, NN);
      rowdot<<<cdiv(NN * 32, 256), B256, 0, stream>>>(xgb, conv_ad + (h * 2 + l) * HID, dB, NN);
      fill_u32<<<cdiv(NN, 256), B256, 0, stream>>>(smax, KEY_NEG_INF, NN);
      gat_alpha_max<<<cdiv(EE, 256), B256, 0, stream>>>(dA, dB, src, dst, alp, smax);
      fill_f32<<<cdiv(NN, 256), B256, 0, stream>>>(sden, 0.f, NN);
      seg_exp_sum<<<cdiv(EE, 256), B256, 0, stream>>>(alp, smax, sden, dst, EE);
      fill_rowbias<<<cdiv(NN * HID, 256), B256, 0, stream>>>(amb, conv_b + (h * 2 + l) * HID, NN);
      seg_scatter<<<cdiv(EE * HID, 256), B256, 0, stream>>>(xgb, src, alp, sden, dst, amb, EE);
      elu_inplace<<<cdiv(NN * HID, 256), B256, 0, stream>>>(amb, NN * HID);
      const size_t gw = (size_t)(h * 3 + l + 1) * 384 * HID;
      gemmF(amb, HID, gwih + gw, HID, gbih + (h * 3 + l + 1) * 384, gib, 384, NN, HID, 384, 0);
      gemmF(xb,  HID, gwhh + gw, HID, gbhh + (h * 3 + l + 1) * 384, ghb, 384, NN, HID, 384, 0);
      gru_fuse<<<cdiv(NN * HID, 256), B256, 0, stream>>>(gib, ghb, xb, NN);
    }

    // ---- molecular attentive readout ----
    fill_f32<<<cdiv(GG * HID, 256), B256, 0, stream>>>(gbuf, 0.f, GG * HID);
    seg_sum_rows<<<cdiv(NN * HID, 256), B256, 0, stream>>>(xb, batch, gbuf, NN);
    relu_inplace<<<cdiv(GG * HID, 256), B256, 0, stream>>>(gbuf, GG * HID);
    gemmF(xb, HID, mws + (size_t)h * HID * HID, HID, nullptr, xsb, HID, NN, HID, HID, 0);
    rowdot<<<cdiv(NN * 32, 256), B256, 0, stream>>>(xsb, mas + h * HID, dA, NN);
    for (int t = 0; t < 3; ++t) {
      gemmF(gbuf, HID, mwd + (size_t)h * HID * HID, HID, nullptr, xgb, HID, GG, HID, HID, 0);
      rowdot<<<cdiv(GG * 32, 256), B256, 0, stream>>>(xgb, mad + h * HID, dG, GG);
      fill_u32<<<cdiv(GG, 256), B256, 0, stream>>>(smax, KEY_NEG_INF, GG);
      mol_alpha_max<<<cdiv(NN, 256), B256, 0, stream>>>(dA, dG, batch, alp, smax);
      fill_f32<<<cdiv(GG, 256), B256, 0, stream>>>(sden, 0.f, GG);
      seg_exp_sum<<<cdiv(NN, 256), B256, 0, stream>>>(alp, smax, sden, batch, NN);
      fill_rowbias<<<cdiv(GG * HID, 256), B256, 0, stream>>>(amb, mbias + h * HID, GG);
      seg_scatter<<<cdiv(NN * HID, 256), B256, 0, stream>>>(xsb, nullptr, alp, sden, batch, amb, NN);
      elu_inplace<<<cdiv(GG * HID, 256), B256, 0, stream>>>(amb, GG * HID);
      gemmF(amb,  HID, mgwih + (size_t)h * 384 * HID, HID, mgbih + h * 384, gib, 384, GG, HID, 384, 0);
      gemmF(gbuf, HID, mgwhh + (size_t)h * 384 * HID, HID, mgbhh + h * 384, ghb, 384, GG, HID, 384, 0);
      gru_fuse<<<cdiv(GG * HID, 256), B256, 0, stream>>>(gib, ghb, gbuf, GG);
    }
    // head output -> cat[:, h*128 : h*128+128]
    gemmF(gbuf, HID, out_w + (size_t)h * HID * HID, HID, out_b + h * HID,
          catb + h * HID, 3 * HID, GG, HID, HID, 0);
  }

  // ---- final: relu(BN(cat @ att_w.T + att_b)) ----
  gemmF(catb, 3 * HID, att_w, 3 * HID, att_b, ftmp, HID, GG, 3 * HID, HID, 0);
  bn(ftmp, OUT, att_bg, att_bb, GG, false);
}
